// ClassicalQLSTM_65481071396753
// MI455X (gfx1250) — matrix operand order repaired and checked by reading, not compiled, not run on hardware
//
#include <hip/hip_runtime.h>
#include <hip/hip_bf16.h>

// ---------------------------------------------------------------------------
// LSTM: T=512 steps, B=64 batch, D=H=1024.
//   Gx = X @ Wx + b   (time-parallel, WMMA bf16, 2x4 register-blocked)
//   per step: gates = h_prev @ Wh + Gx[t]; K-split x4 + LDS reduce + fused
//   sigmoid/tanh cell update.
// ---------------------------------------------------------------------------

#define LT 512
#define LB 64
#define LD 1024
#define LH 1024
#define LN4 (4 * LH)       // 4096 gate columns, order f,i,g,o

typedef __bf16 bf16_t;
typedef __attribute__((ext_vector_type(16))) __bf16 v16bf;
typedef __attribute__((ext_vector_type(8)))  __bf16 v8bf;
typedef __attribute__((ext_vector_type(8)))  float  v8f;

// ---- fragment loaders (wave32 WMMA 16x16x32 bf16 layouts) ------------------
// A (16x32, row-major source, leading dim ld):
//   lane L: row M = L&15 ; lanes 0-15 hold K 0..15, lanes 16-31 hold
//   K 16..31  -> two contiguous 16B loads.
__device__ __forceinline__ v16bf load_frag_a(const bf16_t* __restrict__ tile,
                                             int ld, int lane) {
    const int r  = lane & 15;
    const int kb = (lane >> 4) * 16;
    const bf16_t* q = tile + (size_t)r * ld + kb;
    v8bf lo = *(const v8bf*)(q);
    v8bf hi = *(const v8bf*)(q + 8);
    v16bf out;
#pragma unroll
    for (int i = 0; i < 8; ++i) { out[i] = lo[i]; out[i + 8] = hi[i]; }
    return out;
}

// B (32x16): source is W^T, row-major over N with K contiguous (leading dim ld).
//   lane L: col N = L&15 ; lanes 0-15 hold K 0..15, lanes 16-31 hold K 16..31
//   -> one contiguous 32B load.
__device__ __forceinline__ v16bf load_frag_b(const bf16_t* __restrict__ wt,
                                             int ld, int lane) {
    const int n  = lane & 15;
    const int kb = (lane >> 4) * 16;
    return *(const v16bf*)(wt + (size_t)n * ld + kb);
}

__device__ __forceinline__ float sigmoidf_(float x) {
    return 1.0f / (1.0f + __expf(-x));
}

// ---- weight pack: W_g[(D+H), H] (row-major [in,out]) -> bf16 transposed ----
// WTx[n, k] (n = gate*H + hcol, k in [0,D)) and WTh[n, k] (k in [0,H))
__global__ void __launch_bounds__(256)
pack_weights(const float* __restrict__ Wf, const float* __restrict__ Wi,
             const float* __restrict__ Wg, const float* __restrict__ Wo,
             bf16_t* __restrict__ WTx, bf16_t* __restrict__ WTh) {
    const size_t per_gate = (size_t)(LD + LH) * LH;
    size_t idx = (size_t)blockIdx.x * blockDim.x + threadIdx.x;
    if (idx >= 4 * per_gate) return;
    const int gate = (int)(idx / per_gate);
    const size_t rem = idx % per_gate;
    const int k    = (int)(rem / LH);
    const int hcol = (int)(rem % LH);
    const float* W = (gate == 0) ? Wf : (gate == 1) ? Wi : (gate == 2) ? Wg : Wo;
    const float v = W[(size_t)k * LH + hcol];
    const int n = gate * LH + hcol;
    if (k < LD) WTx[(size_t)n * LD + k] = (bf16_t)v;
    else        WTh[(size_t)n * LH + (k - LD)] = (bf16_t)v;
}

__global__ void __launch_bounds__(256)
convert_x(const float* __restrict__ X, bf16_t* __restrict__ Xb, size_t n) {
    size_t i = (size_t)blockIdx.x * blockDim.x + threadIdx.x;
    if (i < n) Xb[i] = (bf16_t)X[i];
}

__global__ void __launch_bounds__(256)
init_state(bf16_t* __restrict__ h0, float* __restrict__ c0) {
    size_t i = (size_t)blockIdx.x * blockDim.x + threadIdx.x;
    if (i < (size_t)LB * LH) { h0[i] = (bf16_t)0.0f; c0[i] = 0.0f; }
}

// ---- Gx = Xb[T*B, D] @ WTx^T + bias --------------------------------------
// Register-blocked: each wave computes a 32x64 output patch (2 M-tiles x
// 4 N-tiles): per K-step 6 fragment loads feed 8 WMMAs.
__global__ void __launch_bounds__(256)
gemm_gates_x(const bf16_t* __restrict__ Xb, const bf16_t* __restrict__ WTx,
             const float* __restrict__ bf, const float* __restrict__ bi,
             const float* __restrict__ bg, const float* __restrict__ bo,
             float* __restrict__ Gx) {
    const int lane = threadIdx.x & 31;
    const int wid  = blockIdx.x * (blockDim.x >> 5) + (threadIdx.x >> 5);
    const int NG = (LN4 / 16) / 4;           // 64 n-groups (4 tiles each)
    const int mg = wid / NG;                 // 0..1023 (m-groups of 2 tiles)
    const int ng = wid % NG;                 // 0..63
    const bf16_t* A0 = Xb + (size_t)(mg * 32) * LD;
    const bf16_t* A1 = A0 + (size_t)16 * LD;
    const bf16_t* Bb = WTx + (size_t)(ng * 64) * LD;
    v8f acc[2][4] = {};
#pragma unroll 1
    for (int k = 0; k < LD; k += 32) {
        v16bf a0 = load_frag_a(A0 + k, LD, lane);
        v16bf a1 = load_frag_a(A1 + k, LD, lane);
#pragma unroll
        for (int ni = 0; ni < 4; ++ni) {
            v16bf b = load_frag_b(Bb + (size_t)ni * 16 * LD + k, LD, lane);
            acc[0][ni] = __builtin_amdgcn_wmma_f32_16x16x32_bf16(
                false, a0, false, b, (short)0, acc[0][ni], false, false);
            acc[1][ni] = __builtin_amdgcn_wmma_f32_16x16x32_bf16(
                false, a1, false, b, (short)0, acc[1][ni], false, false);
        }
    }
#pragma unroll
    for (int ni = 0; ni < 4; ++ni) {
        const int n_tile = ng * 4 + ni;
        const int gate = n_tile >> 6;        // 64 n-tiles per gate
        const int hcol = ((n_tile & 63) << 4) + (lane & 15);
        const float* bptr = (gate == 0) ? bf : (gate == 1) ? bi
                          : (gate == 2) ? bg : bo;
        const float bias = bptr[hcol];
        const int ncol = n_tile * 16 + (lane & 15);
#pragma unroll
        for (int mi = 0; mi < 2; ++mi) {
            const int mbase = (mg * 2 + mi) * 16 + 8 * (lane >> 4);
#pragma unroll
            for (int r = 0; r < 8; ++r)
                Gx[(size_t)(mbase + r) * LN4 + ncol] = acc[mi][ni][r] + bias;
        }
    }
}

// ---- fused recurrent step --------------------------------------------------
// 256 output units (m_tile 0..3 x nh_tile 0..63), each with 4 gate tiles.
// K split 4-ways -> 1024 waves/step (128 blocks x 8 waves, 2 units/block).
// Partial accumulators reduced through LDS; ksplit==0 wave applies the
// sigmoid/tanh cell update.
__global__ void __launch_bounds__(256)
lstm_step(const bf16_t* __restrict__ hprev, const bf16_t* __restrict__ WTh,
          const float* __restrict__ Gxt, float* __restrict__ c,
          float* __restrict__ hout, bf16_t* __restrict__ hnext) {
    __shared__ float red[2 * 3 * 4 * 256];   // [unit][ksplit-1][gate][lane*8+r]
    const int lane = threadIdx.x & 31;
    const int wave = threadIdx.x >> 5;       // 0..7
    const int unit_local = wave >> 2;        // 0..1
    const int ksplit = wave & 3;             // 0..3
    const int unit = blockIdx.x * 2 + unit_local;  // 0..255
    const int m_tile  = unit >> 6;           // 0..3
    const int nh_tile = unit & 63;           // 0..63
    const bf16_t* Abase = hprev + (size_t)m_tile * 16 * LH;
    const bf16_t* Bbase = WTh + (size_t)(nh_tile * 16) * LH;
    v8f acc[4] = {};
    const int k0 = ksplit * (LH / 4);
#pragma unroll 2
    for (int k = k0; k < k0 + LH / 4; k += 32) {
        v16bf a = load_frag_a(Abase + k, LH, lane);
#pragma unroll
        for (int g = 0; g < 4; ++g) {
            v16bf b = load_frag_b(Bbase + (size_t)g * LH * LH + k, LH, lane);
            acc[g] = __builtin_amdgcn_wmma_f32_16x16x32_bf16(
                false, a, false, b, (short)0, acc[g], false, false);
        }
    }
    if (ksplit != 0) {
#pragma unroll
        for (int g = 0; g < 4; ++g) {
            float* dst = &red[((unit_local * 3 + (ksplit - 1)) * 4 + g) * 256
                              + lane * 8];
#pragma unroll
            for (int r = 0; r < 8; ++r) dst[r] = acc[g][r];
        }
    }
    __syncthreads();
    if (ksplit == 0) {
#pragma unroll
        for (int g = 0; g < 4; ++g) {
#pragma unroll
            for (int ks = 0; ks < 3; ++ks) {
                const float* src = &red[((unit_local * 3 + ks) * 4 + g) * 256
                                        + lane * 8];
#pragma unroll
                for (int r = 0; r < 8; ++r) acc[g][r] += src[r];
            }
        }
        const int hcol  = (nh_tile << 4) + (lane & 15);
        const int mbase = m_tile * 16 + 8 * (lane >> 4);
#pragma unroll
        for (int r = 0; r < 8; ++r) {
            const int batch = mbase + r;
            const size_t gr = (size_t)batch * LN4;
            const float f = sigmoidf_(acc[0][r] + Gxt[gr + 0 * LH + hcol]);
            const float i = sigmoidf_(acc[1][r] + Gxt[gr + 1 * LH + hcol]);
            const float g = tanhf    (acc[2][r] + Gxt[gr + 2 * LH + hcol]);
            const float o = sigmoidf_(acc[3][r] + Gxt[gr + 3 * LH + hcol]);
            const size_t ci = (size_t)batch * LH + hcol;
            const float cn = f * c[ci] + i * g;
            c[ci] = cn;
            const float hv = o * tanhf(cn);
            hout[ci]  = hv;
            hnext[ci] = (bf16_t)hv;
        }
    }
}

__global__ void __launch_bounds__(256)
finalize(const float* __restrict__ hlast, const float* __restrict__ c,
         float* __restrict__ hx_out, float* __restrict__ cx_out) {
    size_t i = (size_t)blockIdx.x * blockDim.x + threadIdx.x;
    if (i < (size_t)LB * LH) { hx_out[i] = hlast[i]; cx_out[i] = c[i]; }
}

// ---------------------------------------------------------------------------
extern "C" void kernel_launch(void* const* d_in, const int* in_sizes, int n_in,
                              void* d_out, int out_size, void* d_ws, size_t ws_size,
                              hipStream_t stream) {
    const float* X  = (const float*)d_in[0];
    const float* Wf = (const float*)d_in[1];
    const float* bf = (const float*)d_in[2];
    const float* Wi = (const float*)d_in[3];
    const float* bi = (const float*)d_in[4];
    const float* Wg = (const float*)d_in[5];
    const float* bg = (const float*)d_in[6];
    const float* Wo = (const float*)d_in[7];
    const float* bo = (const float*)d_in[8];
    float* out = (float*)d_out;          // [T*B*H] stacked, then hx, then cx

    // workspace carve-up
    char* ws = (char*)d_ws;
    size_t off = 0;
    auto carve = [&](size_t bytes) -> char* {
        char* p = ws + off;
        off += (bytes + 255) & ~(size_t)255;
        return p;
    };
    bf16_t* WTx = (bf16_t*)carve((size_t)LN4 * LD * sizeof(bf16_t)); //  8 MB
    bf16_t* WTh = (bf16_t*)carve((size_t)LN4 * LH * sizeof(bf16_t)); //  8 MB
    bf16_t* Xb  = (bf16_t*)carve((size_t)LT * LB * LD * sizeof(bf16_t)); // 64 MB
    float*  Gx  = (float*) carve((size_t)LT * LB * LN4 * sizeof(float)); // 512 MB
    float*  cst = (float*) carve((size_t)LB * LH * sizeof(float));
    bf16_t* hA  = (bf16_t*)carve((size_t)LB * LH * sizeof(bf16_t));
    bf16_t* hB  = (bf16_t*)carve((size_t)LB * LH * sizeof(bf16_t));
    (void)ws_size; (void)in_sizes; (void)n_in; (void)out_size;

    // 1) pack/transpose weights to bf16
    {
        const size_t n = 4ull * (LD + LH) * LH;
        pack_weights<<<(unsigned)((n + 255) / 256), 256, 0, stream>>>(Wf, Wi, Wg, Wo, WTx, WTh);
    }
    // 2) convert inputs to bf16
    {
        const size_t n = (size_t)LT * LB * LD;
        convert_x<<<(unsigned)((n + 255) / 256), 256, 0, stream>>>(X, Xb, n);
    }
    // 3) zero initial state
    init_state<<<(LB * LH + 255) / 256, 256, 0, stream>>>(hA, cst);
    // 4) big time-parallel GEMM: Gx = Xb @ Wx + b (2x4 blocked waves)
    {
        const int m_groups = (LT * LB) / 32;  // 1024
        const int n_groups = (LN4 / 16) / 4;  // 64
        const int waves    = m_groups * n_groups; // 65536
        gemm_gates_x<<<waves / 8, 256, 0, stream>>>(Xb, WTx, bf, bi, bg, bo, Gx);
    }
    // 5) sequential recurrence: 128 blocks x 8 waves per step (K split x4)
    bf16_t* hin = hA;
    bf16_t* hou = hB;
    for (int t = 0; t < LT; ++t) {
        lstm_step<<<128, 256, 0, stream>>>(hin, WTh,
                                           Gx + (size_t)t * LB * LN4,
                                           cst,
                                           out + (size_t)t * LB * LH,
                                           hou);
        bf16_t* tmp = hin; hin = hou; hou = tmp;
    }
    // 6) hx = stacked[T-1], cx = c
    finalize<<<(LB * LH + 255) / 256, 256, 0, stream>>>(
        out + (size_t)(LT - 1) * LB * LH, cst,
        out + (size_t)LT * LB * LH,
        out + (size_t)LT * LB * LH + (size_t)LB * LH);
}
